// T5Attention_11115375362452
// MI455X (gfx1250) — matrix operand order repaired and checked by reading, not compiled, not run on hardware
//
#include <hip/hip_runtime.h>
#include <hip/hip_bf16.h>

// ---------------------------------------------------------------------------
// T5 encoder self-attention for MI455X (gfx1250), wave32 + WMMA bf16 + TDM.
// B=2, S=2048, D=1024, H=16, Dkv=64, INNER=1024, NUM_BUCKETS=32, MAXDIST=128
// ---------------------------------------------------------------------------

typedef __attribute__((ext_vector_type(16))) __bf16 v16bf;
typedef __attribute__((ext_vector_type(8)))  __bf16 v8bf;
typedef __attribute__((ext_vector_type(2)))  __bf16 v2bf;
typedef __attribute__((ext_vector_type(8)))  float  v8f;
typedef __attribute__((ext_vector_type(2)))  float  f32x2;
typedef __attribute__((ext_vector_type(4)))  unsigned int u32x4;
typedef __attribute__((ext_vector_type(8)))  int    i32x8;
typedef __attribute__((ext_vector_type(4)))  int    i32x4;

#define BATCH 2
#define SEQ   2048
#define DMODEL 1024
#define NHEADS 16
#define DKV   64
#define BH    (BATCH * NHEADS)
#define NROWS (BATCH * SEQ)          // 4096
#define NBIAS 4095                   // relative offsets -2047..2047

static __device__ __forceinline__ v8f wmma_bf16(v16bf a, v16bf b, v8f c) {
  return __builtin_amdgcn_wmma_f32_16x16x32_bf16(
      /*neg_a=*/false, a, /*neg_b=*/false, b,
      /*c_mod=*/(short)0, c, /*reuse_a=*/false, /*reuse_b=*/false);
}

// A fragment (16x32 bf16, M=lane&15) from a K-contiguous bf16 row.
// element i <-> K = k0 + 16*half + i; pairs are contiguous.
static __device__ __forceinline__ v16bf load_a_bf16(const unsigned int* row32,
                                                    int k0, int half) {
  v16bf a;
#pragma unroll
  for (int p = 0; p < 8; ++p) {
    int idx = (k0 >> 1) + half * 8 + p;
    v2bf t = __builtin_bit_cast(v2bf, row32[idx]);
    a[2 * p] = t[0];
    a[2 * p + 1] = t[1];
  }
  return a;
}

// A fragment built from an f32 row (on-the-fly bf16 convert).
static __device__ __forceinline__ v16bf load_a_f32(const float* row, int k0,
                                                   int half) {
  v16bf a;
#pragma unroll
  for (int p = 0; p < 8; ++p) {
    int k = k0 + half * 16 + p * 2;
    f32x2 t = *(const f32x2*)(row + k);
    a[2 * p] = (__bf16)t[0];
    a[2 * p + 1] = (__bf16)t[1];
  }
  return a;
}

// B fragment (32x16 bf16, N=lane&15): element i <-> K = 16*half + i.
// p must point at the 16 contiguous bf16 K-values for this lane's column.
static __device__ __forceinline__ v16bf load_b_bf16(const __bf16* p) {
  v8bf lo = __builtin_bit_cast(v8bf, *(const u32x4*)p);
  v8bf hi = __builtin_bit_cast(v8bf, *(const u32x4*)(p + 8));
  v16bf b;
#pragma unroll
  for (int i = 0; i < 8; ++i) {
    b[i] = lo[i];
    b[i + 8] = hi[i];
  }
  return b;
}

// ---------------------------------------------------------------------------
// Tensor Data Mover helpers (CDNA5 ISA, D# groups 0/1).
// 2D tile of 2-byte elements: LDS gets tile_d1 x tile_d0 row-major compact.
// tile_d1 == 0 -> 1D transfer of tile_d0 elements.
// ---------------------------------------------------------------------------
static __device__ __forceinline__ unsigned lds_byte_offset(const void* p) {
  return (unsigned)(unsigned long long)(
      __attribute__((address_space(3))) const void*)p;
}

static __device__ __forceinline__ void tdm_load_bf16(
    unsigned lds_off, const void* gptr, unsigned tile_d0, unsigned tile_d1,
    unsigned tensor_d0, unsigned tensor_d1, unsigned long long stride0) {
  unsigned long long ga = (unsigned long long)gptr;
  u32x4 g0;
  g0[0] = 1u;                                   // count=1, user mode
  g0[1] = lds_off;                              // lds_addr (bytes)
  g0[2] = (unsigned)ga;                         // global_addr[31:0]
  g0[3] = (unsigned)((ga >> 32) & 0x01FFFFFFu)  // global_addr[56:32]
          | (2u << 30);                         // type=2 ("image")
  i32x8 g1;
  g1[0] = (int)(1u << 16);                      // data_size=2B, mask=0
  g1[1] = (int)((tensor_d0 & 0xFFFFu) << 16);   // tensor_dim0[15:0]
  g1[2] = (int)((tensor_d0 >> 16) | ((tensor_d1 & 0xFFFFu) << 16));
  g1[3] = (int)((tensor_d1 >> 16) | (tile_d0 << 16));   // tile_dim0
  g1[4] = (int)(tile_d1 & 0xFFFFu);             // tile_dim1 (0 => 1D)
  g1[5] = (int)(unsigned)(stride0 & 0xFFFFFFFFull);     // dim0_stride lo32
  g1[6] = (int)(unsigned)((stride0 >> 32) & 0xFFFFull); // dim0_stride hi16
  g1[7] = 0;
  i32x4 z4 = {};
#if defined(__clang_major__) && __clang_major__ >= 23
  i32x8 z8 = {};
  __builtin_amdgcn_tensor_load_to_lds(g0, g1, z4, z4, z8, 0);
#else
  __builtin_amdgcn_tensor_load_to_lds(g0, g1, z4, z4, 0);
#endif
}

// ---------------------------------------------------------------------------
// Kernel 1: transpose+convert weights to bf16, K-contiguous (N-major).
// WT  : [3072][1024]  rows n in {wq|wk|wv columns}, cols = K
// WoT : [1024][1024]
// ---------------------------------------------------------------------------
__global__ void prep_weights(const float* __restrict__ wq,
                             const float* __restrict__ wk,
                             const float* __restrict__ wv,
                             const float* __restrict__ wo,
                             __bf16* __restrict__ WT,
                             __bf16* __restrict__ WoT) {
  int idx = blockIdx.x * 256 + threadIdx.x;  // 0 .. 4194303
  if (idx < 3 * 1024 * 1024) {
    int n = idx >> 10, k = idx & 1023;
    int mat = n >> 10, c = n & 1023;
    const float* w = (mat == 0) ? wq : (mat == 1) ? wk : wv;
    WT[idx] = (__bf16)w[k * 1024 + c];
  } else {
    int j = idx - 3 * 1024 * 1024;
    int n = j >> 10, k = j & 1023;
    WoT[j] = (__bf16)wo[k * 1024 + n];
  }
}

// ---------------------------------------------------------------------------
// Kernel 2: relative-position bias table  biasT[h][rel+2047]
// ---------------------------------------------------------------------------
__global__ void bias_table(const float* __restrict__ rel_bias,
                           float* __restrict__ biasT) {
  int tid = blockIdx.x * 256 + threadIdx.x;
  if (tid >= NHEADS * NBIAS) return;
  int h = tid / NBIAS;
  int rel = tid % NBIAS - 2047;
  int bucket = (rel > 0) ? 16 : 0;
  int rp = (rel < 0) ? -rel : rel;
  if (rp < 8) {
    bucket += rp;
  } else {
    int lg = 8 + (int)(logf((float)rp * 0.125f) * (8.0f / logf(16.0f)));
    bucket += (lg < 15) ? lg : 15;
  }
  biasT[tid] = rel_bias[bucket * NHEADS + h];
}

// ---------------------------------------------------------------------------
// Kernel 3: fused QKV projection.  X[4096x1024] @ WT' -> Q,K bf16 [BH,S,64],
// V stored transposed: VT bf16 [BH,64,S].  Wave owns a 16x64 strip.
// Double-buffered fragment prefetch to keep loadcnt ahead of the WMMAs.
// ---------------------------------------------------------------------------
__global__ __launch_bounds__(256) void qkv_proj(const float* __restrict__ X,
                                                const __bf16* __restrict__ WT,
                                                __bf16* __restrict__ Qb,
                                                __bf16* __restrict__ Kb,
                                                __bf16* __restrict__ VTb) {
  int lane = threadIdx.x & 31, wave = threadIdx.x >> 5;
  int half = lane >> 4, l15 = lane & 15;
  int m0 = blockIdx.x * 16;
  int nbase = blockIdx.y * 512 + wave * 64;

  v8f acc[4] = {};
  const float* xrow = X + (m0 + l15) * DMODEL;
  const __bf16* wrow = WT + (nbase + l15) * DMODEL + half * 16;

  v16bf a_cur = load_a_f32(xrow, 0, half);
  v16bf b_cur[4];
#pragma unroll
  for (int f = 0; f < 4; ++f) b_cur[f] = load_b_bf16(wrow + f * 16 * DMODEL);

#pragma unroll 2
  for (int k0 = 0; k0 < DMODEL; k0 += 32) {
    int kn = (k0 + 32 < DMODEL) ? k0 + 32 : k0;  // clamp (dead last prefetch)
    v16bf a_nxt = load_a_f32(xrow, kn, half);
    v16bf b_nxt[4];
#pragma unroll
    for (int f = 0; f < 4; ++f)
      b_nxt[f] = load_b_bf16(wrow + f * 16 * DMODEL + kn);
#pragma unroll
    for (int f = 0; f < 4; ++f) acc[f] = wmma_bf16(a_cur, b_cur[f], acc[f]);
    a_cur = a_nxt;
#pragma unroll
    for (int f = 0; f < 4; ++f) b_cur[f] = b_nxt[f];
  }

  int b = m0 >> 11, s0 = m0 & 2047;
  int mat = nbase >> 10;               // uniform per wave
  int c0 = nbase & 1023;
  int h = c0 >> 6, bh = b * NHEADS + h;
#pragma unroll
  for (int f = 0; f < 4; ++f) {
    int d = (c0 & 63) + f * 16 + l15;  // 0..63 within the head
#pragma unroll
    for (int r = 0; r < 8; ++r) {
      int s = s0 + r + 8 * half;
      __bf16 v = (__bf16)acc[f][r];
      if (mat == 0)      Qb[(bh * SEQ + s) * DKV + d] = v;
      else if (mat == 1) Kb[(bh * SEQ + s) * DKV + d] = v;
      else               VTb[(bh * DKV + d) * SEQ + s] = v;
    }
  }
}

// ---------------------------------------------------------------------------
// Kernel 4: flash attention.  One wave per (bh, 16-query tile).
// K/V tiles staged into LDS by the Tensor Data Mover, double-buffered with
// s_wait_tensorcnt.  Online softmax; P goes D-layout -> LDS -> A-layout.
// ---------------------------------------------------------------------------
__global__ __launch_bounds__(32) void attn(const __bf16* __restrict__ Qb,
                                           const __bf16* __restrict__ Kb,
                                           const __bf16* __restrict__ VTb,
                                           const float* __restrict__ biasT,
                                           __bf16* __restrict__ attb) {
  __shared__ __align__(16) __bf16 lds_k[2][32 * DKV];  // [key][d]   4KB x2
  __shared__ __align__(16) __bf16 lds_v[2][DKV * 32];  // [d][key]   4KB x2
  __shared__ __align__(16) __bf16 lds_p[16 * 32];      // P tile     1KB

  int lane = threadIdx.x, half = lane >> 4, l15 = lane & 15;
  int bh = blockIdx.x, b = bh >> 4, h = bh & 15;
  int q0 = blockIdx.y * 16;

  const unsigned int* qrow =
      (const unsigned int*)(Qb + (bh * SEQ + q0 + l15) * DKV);
  v16bf qf0 = load_a_bf16(qrow, 0, half);
  v16bf qf1 = load_a_bf16(qrow, 32, half);

  float mrow[8], lrow[8];
  v8f O[4] = {};
#pragma unroll
  for (int r = 0; r < 8; ++r) { mrow[r] = -3.0e38f; lrow[r] = 0.0f; }

  // btab[key - r] == bias for rel = key - (q0 + r + 8*half)
  const float* btab = biasT + h * NBIAS + 2047 - (q0 + 8 * half);

  const __bf16* kbase_ptr = Kb + bh * SEQ * DKV;
  const __bf16* vbase_ptr = VTb + bh * DKV * SEQ;

  // TDM prefetch of tile kb into buffer buf: K tile is 32x64 contiguous (1D),
  // V tile is 64 rows x 32 cols with row stride SEQ.
  // Preload first tiles.
  tdm_load_bf16(lds_byte_offset(lds_k[0]), kbase_ptr, 2048, 0, 2048, 0, 2048);
  tdm_load_bf16(lds_byte_offset(lds_v[0]), vbase_ptr, 32, DKV, 32, DKV, SEQ);

  int cur = 0;
  for (int kb = 0; kb < SEQ; kb += 32) {
    if (kb + 32 < SEQ) {
      tdm_load_bf16(lds_byte_offset(lds_k[cur ^ 1]),
                    kbase_ptr + (kb + 32) * DKV, 2048, 0, 2048, 0, 2048);
      tdm_load_bf16(lds_byte_offset(lds_v[cur ^ 1]), vbase_ptr + kb + 32, 32,
                    DKV, 32, DKV, SEQ);
      __builtin_amdgcn_s_wait_tensorcnt(2);  // current pair complete
    } else {
      __builtin_amdgcn_s_wait_tensorcnt(0);
    }

    v8f sf[2];
#pragma unroll
    for (int kt = 0; kt < 2; ++kt) {
      v8f s = {};
      const __bf16* krow = lds_k[cur] + (kt * 16 + l15) * DKV;
      s = wmma_bf16(qf0, load_b_bf16(krow + half * 16), s);
      s = wmma_bf16(qf1, load_b_bf16(krow + 32 + half * 16), s);
      int key = kb + kt * 16 + l15;
#pragma unroll
      for (int r = 0; r < 8; ++r) s[r] += btab[key - r];
      sf[kt] = s;
    }

    float scale[8];
#pragma unroll
    for (int r = 0; r < 8; ++r) {
      float c = fmaxf(sf[0][r], sf[1][r]);
      c = fmaxf(c, __shfl_xor(c, 1));
      c = fmaxf(c, __shfl_xor(c, 2));
      c = fmaxf(c, __shfl_xor(c, 4));
      c = fmaxf(c, __shfl_xor(c, 8));
      float mn = fmaxf(mrow[r], c);
      scale[r] = __expf(mrow[r] - mn);
      float p0 = __expf(sf[0][r] - mn);
      float p1 = __expf(sf[1][r] - mn);
      sf[0][r] = p0;
      sf[1][r] = p1;
      float rs = p0 + p1;
      rs += __shfl_xor(rs, 1);
      rs += __shfl_xor(rs, 2);
      rs += __shfl_xor(rs, 4);
      rs += __shfl_xor(rs, 8);
      lrow[r] = lrow[r] * scale[r] + rs;
      mrow[r] = mn;
    }
#pragma unroll
    for (int dt = 0; dt < 4; ++dt)
#pragma unroll
      for (int r = 0; r < 8; ++r) O[dt][r] *= scale[r];

    __syncthreads();  // guard vs previous iteration's LDS P reads
#pragma unroll
    for (int r = 0; r < 8; ++r) {
      lds_p[(r + 8 * half) * 32 + l15] = (__bf16)sf[0][r];
      lds_p[(r + 8 * half) * 32 + 16 + l15] = (__bf16)sf[1][r];
    }
    __syncthreads();
    v16bf pa = load_a_bf16((const unsigned int*)lds_p + l15 * 16, 0, half);
#pragma unroll
    for (int dt = 0; dt < 4; ++dt) {
      const __bf16* vp = lds_v[cur] + (dt * 16 + l15) * 32 + half * 16;
      O[dt] = wmma_bf16(pa, load_b_bf16(vp), O[dt]);
    }
    cur ^= 1;
  }

#pragma unroll
  for (int dt = 0; dt < 4; ++dt)
#pragma unroll
    for (int r = 0; r < 8; ++r) {
      float v = O[dt][r] / lrow[r];
      int row = b * SEQ + q0 + r + 8 * half;
      attb[row * 1024 + h * DKV + dt * 16 + l15] = (__bf16)v;
    }
}

// ---------------------------------------------------------------------------
// Kernel 5: output projection  attb[4096x1024] @ Wo -> out f32 [4096x1024]
// ---------------------------------------------------------------------------
__global__ __launch_bounds__(256) void out_proj(const __bf16* __restrict__ attb,
                                                const __bf16* __restrict__ WoT,
                                                float* __restrict__ out) {
  int lane = threadIdx.x & 31, wave = threadIdx.x >> 5;
  int half = lane >> 4, l15 = lane & 15;
  int m0 = blockIdx.x * 16;
  int nbase = blockIdx.y * 512 + wave * 64;

  v8f acc[4] = {};
  const unsigned int* arow = (const unsigned int*)(attb + (m0 + l15) * 1024);
  const __bf16* wrow = WoT + (nbase + l15) * 1024 + half * 16;

  v16bf a_cur = load_a_bf16(arow, 0, half);
  v16bf b_cur[4];
#pragma unroll
  for (int f = 0; f < 4; ++f) b_cur[f] = load_b_bf16(wrow + f * 16 * 1024);

#pragma unroll 2
  for (int k0 = 0; k0 < 1024; k0 += 32) {
    int kn = (k0 + 32 < 1024) ? k0 + 32 : k0;
    v16bf a_nxt = load_a_bf16(arow, kn, half);
    v16bf b_nxt[4];
#pragma unroll
    for (int f = 0; f < 4; ++f)
      b_nxt[f] = load_b_bf16(wrow + f * 16 * 1024 + kn);
#pragma unroll
    for (int f = 0; f < 4; ++f) acc[f] = wmma_bf16(a_cur, b_cur[f], acc[f]);
    a_cur = a_nxt;
#pragma unroll
    for (int f = 0; f < 4; ++f) b_cur[f] = b_nxt[f];
  }
#pragma unroll
  for (int f = 0; f < 4; ++f)
#pragma unroll
    for (int r = 0; r < 8; ++r)
      out[(m0 + r + 8 * half) * 1024 + nbase + f * 16 + l15] = acc[f][r];
}

// ---------------------------------------------------------------------------
extern "C" void kernel_launch(void* const* d_in, const int* in_sizes, int n_in,
                              void* d_out, int out_size, void* d_ws,
                              size_t ws_size, hipStream_t stream) {
  const float* hidden = (const float*)d_in[0];
  const float* wq = (const float*)d_in[1];
  const float* wk = (const float*)d_in[2];
  const float* wv = (const float*)d_in[3];
  const float* wo = (const float*)d_in[4];
  const float* rel_bias = (const float*)d_in[5];
  float* out = (float*)d_out;

  char* p = (char*)d_ws;
  __bf16* WT = (__bf16*)p;   p += (size_t)3072 * 1024 * 2;
  __bf16* WoT = (__bf16*)p;  p += (size_t)1024 * 1024 * 2;
  __bf16* Qb = (__bf16*)p;   p += (size_t)BH * SEQ * DKV * 2;
  __bf16* Kb = (__bf16*)p;   p += (size_t)BH * SEQ * DKV * 2;
  __bf16* VTb = (__bf16*)p;  p += (size_t)BH * SEQ * DKV * 2;
  __bf16* attb = (__bf16*)p; p += (size_t)NROWS * 1024 * 2;
  float* biasT = (float*)p;  p += (size_t)NHEADS * NBIAS * 4;

  prep_weights<<<(4 * 1024 * 1024) / 256, 256, 0, stream>>>(wq, wk, wv, wo, WT,
                                                            WoT);
  bias_table<<<(NHEADS * NBIAS + 255) / 256, 256, 0, stream>>>(rel_bias, biasT);
  qkv_proj<<<dim3(NROWS / 16, 6), 256, 0, stream>>>(hidden, WT, Qb, Kb, VTb);
  attn<<<dim3(BH, SEQ / 16), 32, 0, stream>>>(Qb, Kb, VTb, biasT, attb);
  out_proj<<<dim3(NROWS / 16, 2), 256, 0, stream>>>(attb, WoT, out);
}
